// ADCompositeNet_23931557773813
// MI455X (gfx1250) — matrix operand order repaired and checked by reading, not compiled, not run on hardware
//
#include <hip/hip_runtime.h>

// ---------------------------------------------------------------------------
// ADCompositeNet for MI455X (gfx1250, wave32, WMMA + TDM)
// ---------------------------------------------------------------------------

typedef __attribute__((ext_vector_type(16))) _Float16 v16h;
typedef __attribute__((ext_vector_type(8)))  float    v8f;
typedef unsigned int u32x4 __attribute__((ext_vector_type(4)));
typedef int          i32x4 __attribute__((ext_vector_type(4)));
typedef int          i32x8 __attribute__((ext_vector_type(8)));

#define LEAK  0.1f
#define KNN   32
#define NPTS  16384
#define BATCH 32
#define HID   32
#define NCEN  16

__device__ __forceinline__ float leakyf(float x) { return x >= 0.0f ? x : LEAK * x; }

__device__ __forceinline__ v8f wmma_f16(v16h a, v16h b, v8f c) {
  // D = A(16x32 f16) * B(32x16 f16) + C(16x16 f32)
  return __builtin_amdgcn_wmma_f32_16x16x32_f16(false, a, false, b, (short)0, c,
                                                false, false);
}

// ---------------------------------------------------------------------------
// Tensor Data Mover: 1-D contiguous copy of `nelem` f32 from global to LDS.
// D# per CDNA5 ISA:
//   group0: count=1 | lds_addr[63:32] | global_addr[120:64] | type=2[127:126]
//   group1: data_size=2(4B)@[17:16], tensor_dim0@[79:48], tile_dim0@[127:112],
//           tensor_dim0_stride@[207:160]
// Tracked by TENSORcnt; EXEC ignored (issue from one lane, whole-wave wait).
// ---------------------------------------------------------------------------
#if __has_builtin(__builtin_amdgcn_tensor_load_to_lds) && \
    __has_builtin(__builtin_amdgcn_s_wait_tensorcnt)
#define USE_TDM 1
#else
#define USE_TDM 0
#endif

#if USE_TDM
__device__ __forceinline__ void tdm_copy_f32_to_lds(unsigned lds_off,
                                                    const float* gsrc,
                                                    int nelem) {
  const unsigned long long ga = (unsigned long long)gsrc;
  u32x4 g0;
  g0[0] = 1u;                                     // count=1 (valid descriptor)
  g0[1] = lds_off;                                // lds_addr (bytes)
  g0[2] = (unsigned)(ga & 0xffffffffu);           // global_addr[31:0]
  g0[3] = (unsigned)((ga >> 32) & 0x01ffffffu)    // global_addr[56:32]
          | (2u << 30);                           // type=2 ("image")
  i32x8 g1 = {};
  g1[0] = (int)(2u << 16);                        // data_size=2 (4 bytes)
  g1[1] = (int)((unsigned)(nelem & 0xffff) << 16);// tensor_dim0[15:0]  @bit48
  g1[2] = (int)((unsigned)nelem >> 16);           // tensor_dim0[31:16] @bit64
  g1[3] = (int)((unsigned)(nelem & 0xffff) << 16);// tile_dim0 @bit112
  g1[5] = nelem;                                  // tensor_dim0_stride @bit160
  const i32x4 z4 = {};
#if defined(__clang_major__) && (__clang_major__ >= 23)
  const i32x8 z8 = {};
  __builtin_amdgcn_tensor_load_to_lds(g0, g1, z4, z4, z8, 0);
#else
  __builtin_amdgcn_tensor_load_to_lds(g0, g1, z4, z4, 0);
#endif
}
#endif

// ---------------------------------------------------------------------------
// Kernel 1: kNN (K=32) of the first n_q points among the first n_pts points.
// One workgroup per batch; points DMA'd through LDS in 2048-pt chunks via the
// Tensor Data Mover. Unsorted top-K + running max (equivalent to top_k since
// downstream sums over neighbors; max distance feeds normalization).
// ---------------------------------------------------------------------------
#define CHUNK 2048

__global__ __launch_bounds__(128) void knn_kernel(
    const float* __restrict__ pts,   // (BATCH, NPTS, 3)
    int n_pts, int n_q,
    int*   __restrict__ out_idx,     // (BATCH*n_q, KNN)
    float* __restrict__ out_invmaxd) // (BATCH*n_q)
{
  __shared__ float sp[CHUNK * 3];    // AoS x,y,z — matches contiguous DMA
  __shared__ float bestd[128][KNN];
  __shared__ int   besti[128][KNN];

  const int b = blockIdx.x;
  const int t = threadIdx.x;
  const float* P = pts + (size_t)b * NPTS * 3;
  const bool active = t < n_q;

  float qx = 0.f, qy = 0.f, qz = 0.f;
  if (active) { qx = P[t*3+0]; qy = P[t*3+1]; qz = P[t*3+2]; }

  float worst = 3.4e38f;
  int   wslot = 0;
  int   cnt   = 0;

  for (int base = 0; base < n_pts; base += CHUNK) {
    const int cn = (n_pts - base) < CHUNK ? (n_pts - base) : CHUNK;
    __syncthreads();
#if USE_TDM
    if (t == 0) {
      // flat->LDS: low 32 bits of the flat pointer are the LDS byte offset
      const unsigned lds_off = (unsigned)(unsigned long long)(const void*)sp;
      tdm_copy_f32_to_lds(lds_off, P + (size_t)base * 3, cn * 3);
      __builtin_amdgcn_s_wait_tensorcnt(0);
    }
#else
    for (int i = t; i < cn * 3; i += 128) sp[i] = P[(size_t)base * 3 + i];
#endif
    if (base + CHUNK < n_pts)  // prefetch next chunk -> global_prefetch_b8
      __builtin_prefetch(&P[(size_t)(base + CHUNK) * 3 + t * 3], 0, 0);
    __syncthreads();

    if (active) {
      for (int i = 0; i < cn; ++i) {
        const float dx = sp[i*3+0] - qx;
        const float dy = sp[i*3+1] - qy;
        const float dz = sp[i*3+2] - qz;
        const float d  = dx*dx + dy*dy + dz*dz;
        if (cnt < KNN) {
          bestd[t][cnt] = d; besti[t][cnt] = base + i; ++cnt;
          if (cnt == KNN) {
            float w = -1.f; int s = 0;
            #pragma unroll
            for (int k = 0; k < KNN; ++k)
              if (bestd[t][k] > w) { w = bestd[t][k]; s = k; }
            worst = w; wslot = s;
          }
        } else if (d < worst) {
          bestd[t][wslot] = d; besti[t][wslot] = base + i;
          float w = -1.f; int s = 0;
          #pragma unroll
          for (int k = 0; k < KNN; ++k)
            if (bestd[t][k] > w) { w = bestd[t][k]; s = k; }
          worst = w; wslot = s;
        }
      }
    }
  }

  if (active) {
    const int g = b * n_q + t;
    #pragma unroll
    for (int k = 0; k < KNN; ++k) out_idx[(size_t)g * KNN + k] = besti[t][k];
    out_invmaxd[g] = 1.0f / (sqrtf(fmaxf(worst, 0.0f)) + 1e-6f);
  }
}

// ---------------------------------------------------------------------------
// Kernel 2: geometric MLP + aggregation, one wave32 per query.
//   rel(32x3) --wmma--> h(32x32) --wmma--> w(32x16)/K --wmma--> agg^T(16xC)
// agg stored flattened (m*C + c) per query: the A-matrix of the semantic GEMM.
// ---------------------------------------------------------------------------
__global__ __launch_bounds__(256) void geo_mlp_kernel(
    const float* __restrict__ pts,   // (BATCH, NPTS, 3)
    const float* __restrict__ feat,  // (BATCH, n_src, C) f32
    const int*   __restrict__ idx,   // (BATCH*n_q, KNN)
    const float* __restrict__ invm,  // (BATCH*n_q)
    const float* __restrict__ ws1,   // (3, HID)
    const float* __restrict__ bs1,   // (HID)
    const float* __restrict__ ws2,   // (HID, NCEN)
    float* __restrict__ agg,         // (BATCH*n_q, NCEN*C)
    int n_q, int n_src, int C)
{
  __shared__ _Float16 relh[8][KNN * 4];
  __shared__ _Float16 hbuf[8][KNN * HID];
  __shared__ _Float16 wbuf[8][KNN * NCEN];
  __shared__ int      sidx[8][KNN];

  const int slot = threadIdx.x >> 5;
  const int lane = threadIdx.x & 31;
  const int g    = blockIdx.x * 8 + slot;     // query id (grids sized exactly)
  const int b    = g / n_q;
  const int o    = g % n_q;

  const float* P = pts + (size_t)b * NPTS * 3;
  const float qx = P[o*3+0], qy = P[o*3+1], qz = P[o*3+2];
  const float im = invm[g];

  // lane k handles neighbor k
  {
    const int k  = lane;
    const int ig = idx[(size_t)g * KNN + k];
    sidx[slot][k] = ig;
    relh[slot][k*4+0] = (_Float16)((P[ig*3+0] - qx) * im);
    relh[slot][k*4+1] = (_Float16)((P[ig*3+1] - qy) * im);
    relh[slot][k*4+2] = (_Float16)((P[ig*3+2] - qz) * im);
    relh[slot][k*4+3] = (_Float16)0.0f;
  }
  __syncthreads();

  const int m0    = lane & 15;
  const int koffA = (lane < 16) ? 0 : 16;
  const int koffB = (lane < 16) ? 0 : 16;
  const int hi8   = (lane >= 16) ? 8 : 0;
  const int nB    = lane & 15;

  // ---- h = relu(rel @ ws1 + bs1) : M=32 (neighbors), K=3 (pad32), N=32 ----
  v16h a0, a1;
  #pragma unroll
  for (int e = 0; e < 16; ++e) {
    const int kk = koffA + e;
    a0[e] = (kk < 4) ? relh[slot][m0 * 4 + kk]        : (_Float16)0.0f;
    a1[e] = (kk < 4) ? relh[slot][(m0 + 16) * 4 + kk] : (_Float16)0.0f;
  }
  v16h bw1a, bw1b;
  #pragma unroll
  for (int e = 0; e < 16; ++e) {
    const int kk = koffB + e;
    bw1a[e] = (kk < 3) ? (_Float16)ws1[kk * HID + nB]      : (_Float16)0.0f;
    bw1b[e] = (kk < 3) ? (_Float16)ws1[kk * HID + 16 + nB] : (_Float16)0.0f;
  }
  v8f c00 = {}, c01 = {}, c10 = {}, c11 = {};
  c00 = wmma_f16(a0, bw1a, c00);
  c01 = wmma_f16(a0, bw1b, c01);
  c10 = wmma_f16(a1, bw1a, c10);
  c11 = wmma_f16(a1, bw1b, c11);

  const float bias0 = bs1[nB], bias1 = bs1[16 + nB];
  #pragma unroll
  for (int r = 0; r < 8; ++r) {
    const int m = r + hi8;
    hbuf[slot][m * HID + nB]             = (_Float16)fmaxf(c00[r] + bias0, 0.f);
    hbuf[slot][m * HID + 16 + nB]        = (_Float16)fmaxf(c01[r] + bias1, 0.f);
    hbuf[slot][(m + 16) * HID + nB]      = (_Float16)fmaxf(c10[r] + bias0, 0.f);
    hbuf[slot][(m + 16) * HID + 16 + nB] = (_Float16)fmaxf(c11[r] + bias1, 0.f);
  }
  __syncthreads();

  // ---- w = (h @ ws2) / K : M=32, K=32, N=16 ----
  v16h ha0, ha1, bw2;
  #pragma unroll
  for (int e = 0; e < 16; ++e) {
    const int kk = koffA + e;
    ha0[e] = hbuf[slot][m0 * HID + kk];
    ha1[e] = hbuf[slot][(m0 + 16) * HID + kk];
    bw2[e] = (_Float16)ws2[(koffB + e) * NCEN + nB];
  }
  v8f cw0 = {}, cw1 = {};
  cw0 = wmma_f16(ha0, bw2, cw0);
  cw1 = wmma_f16(ha1, bw2, cw1);
  #pragma unroll
  for (int r = 0; r < 8; ++r) {
    const int m = r + hi8;
    wbuf[slot][m * NCEN + nB]        = (_Float16)(cw0[r] * (1.0f / KNN));
    wbuf[slot][(m + 16) * NCEN + nB] = (_Float16)(cw1[r] * (1.0f / KNN));
  }
  __syncthreads();

  // ---- agg^T = w^T(16x32) @ nx(32xC) ----
  v16h aw;
  #pragma unroll
  for (int e = 0; e < 16; ++e) {
    const int kk = koffA + e;
    aw[e] = wbuf[slot][kk * NCEN + m0];
  }
  const size_t fb = (size_t)b * n_src * C;
  const int ct_n = (C + 15) >> 4;
  for (int ct = 0; ct < ct_n; ++ct) {
    const int c = ct * 16 + nB;
    v16h bx;
    #pragma unroll
    for (int e = 0; e < 16; ++e) {
      const int kk = koffB + e;  // neighbor index
      bx[e] = (c < C) ? (_Float16)feat[fb + (size_t)sidx[slot][kk] * C + c]
                      : (_Float16)0.0f;
    }
    v8f cg = {};
    cg = wmma_f16(aw, bx, cg);
    if (c < C) {
      #pragma unroll
      for (int r = 0; r < 8; ++r) {
        const int m = r + hi8;
        agg[(size_t)g * (NCEN * C) + (size_t)m * C + c] = cg[r];
      }
    }
  }
}

// ---------------------------------------------------------------------------
// Kernel 3: generic WMMA GEMM  Out(MxN) = act(A(MxK) @ W(KxN) + bias)
// A,W f32 in memory, f16 WMMA with f32 accumulate. One wave per 16x16 tile.
// ---------------------------------------------------------------------------
__global__ __launch_bounds__(256) void gemm_kernel(
    const float* __restrict__ A, const float* __restrict__ W,
    const float* __restrict__ bias, float* __restrict__ Out,
    int M, int N, int K, int apply_leaky, int leaky_a)
{
  const int lane   = threadIdx.x & 31;
  const int wid    = (blockIdx.x * blockDim.x + threadIdx.x) >> 5;
  const int ntiles = (N + 15) >> 4;
  const int mtiles = (M + 15) >> 4;
  if (wid >= mtiles * ntiles) return;    // wave-uniform
  const int mt = wid / ntiles;
  const int nt = wid % ntiles;

  const int nB    = lane & 15;
  const int mrow  = mt * 16 + nB;        // A row for this lane's fragment slice
  const int ncol  = nt * 16 + nB;
  const int koffA = (lane < 16) ? 0 : 16;
  const int koffB = (lane < 16) ? 0 : 16;
  const int hi8   = (lane >= 16) ? 8 : 0;

  v8f acc = {};
  #pragma unroll 2
  for (int kb = 0; kb < K; kb += 32) {
    v16h a, bf;
    #pragma unroll
    for (int e = 0; e < 16; ++e) {
      const int ka = kb + koffA + e;
      float av = (ka < K) ? A[(size_t)mrow * K + ka] : 0.0f;
      if (leaky_a) av = leakyf(av);
      a[e] = (_Float16)av;
      const int kw = kb + koffB + e;
      bf[e] = (kw < K && ncol < N) ? (_Float16)W[(size_t)kw * N + ncol]
                                   : (_Float16)0.0f;
    }
    acc = wmma_f16(a, bf, acc);
  }

  const float bv = (bias && ncol < N) ? bias[ncol] : 0.0f;
  if (ncol < N) {
    #pragma unroll
    for (int r = 0; r < 8; ++r) {
      const int m = mt * 16 + r + hi8;
      float v = acc[r] + bv;
      if (apply_leaky) v = leakyf(v);
      Out[(size_t)m * N + ncol] = v;
    }
  }
}

// ---------------------------------------------------------------------------
// Kernel 4: batchnorm over (Bn, Cc) + leaky
// ---------------------------------------------------------------------------
__global__ void bn_leaky_kernel(const float* __restrict__ x,
                                float* __restrict__ out, int Bn, int Cc)
{
  const int c = blockIdx.x * blockDim.x + threadIdx.x;
  if (c >= Cc) return;
  float m = 0.f;
  for (int i = 0; i < Bn; ++i) m += x[(size_t)i * Cc + c];
  m /= (float)Bn;
  float v = 0.f;
  for (int i = 0; i < Bn; ++i) {
    const float d = x[(size_t)i * Cc + c] - m;
    v += d * d;
  }
  v /= (float)Bn;
  const float inv = rsqrtf(v + 1e-5f);
  for (int i = 0; i < Bn; ++i)
    out[(size_t)i * Cc + c] = leakyf((x[(size_t)i * Cc + c] - m) * inv);
}

// ---------------------------------------------------------------------------
extern "C" void kernel_launch(void* const* d_in, const int* in_sizes, int n_in,
                              void* d_out, int out_size, void* d_ws, size_t ws_size,
                              hipStream_t stream)
{
  const float* x      = (const float*)d_in[0];   // (32,16384,3)
  const float* pts    = (const float*)d_in[1];   // (32,16384,3)
  const float* ws1_1  = (const float*)d_in[2];
  const float* bs1_1  = (const float*)d_in[3];
  const float* ws2_1  = (const float*)d_in[4];
  const float* wsem1  = (const float*)d_in[5];
  const float* bsem1  = (const float*)d_in[6];
  const float* ws1_3  = (const float*)d_in[7];
  const float* bs1_3  = (const float*)d_in[8];
  const float* ws2_3  = (const float*)d_in[9];
  const float* wsem3  = (const float*)d_in[10];
  const float* bsem3  = (const float*)d_in[11];
  const float* ws1_4  = (const float*)d_in[12];
  const float* bs1_4  = (const float*)d_in[13];
  const float* ws2_4  = (const float*)d_in[14];
  const float* wsem4  = (const float*)d_in[15];
  const float* bsem4  = (const float*)d_in[16];
  const float* wfco   = (const float*)d_in[17];  // (384,128)
  const float* wfco2  = (const float*)d_in[18];  // (128,20)
  float* out = (float*)d_out;

  // workspace carve-up (256B aligned)
  char* w = (char*)d_ws;
  size_t off = 0;
  auto carve = [&](size_t bytes) { void* p = w + off; off = (off + bytes + 255) & ~(size_t)255; return p; };
  int*   idx1 = (int*)  carve((size_t)BATCH * 128 * KNN * 4);
  float* inv1 = (float*)carve((size_t)BATCH * 128 * 4);
  float* agg1 = (float*)carve((size_t)4096 * 48 * 4);
  float* x1   = (float*)carve((size_t)4096 * 64 * 4);
  int*   idx3 = (int*)  carve((size_t)BATCH * 32 * KNN * 4);
  float* inv3 = (float*)carve((size_t)BATCH * 32 * 4);
  float* agg3 = (float*)carve((size_t)1024 * 1024 * 4);
  float* x3   = (float*)carve((size_t)1024 * 192 * 4);
  int*   idx4 = (int*)  carve((size_t)BATCH * 1 * KNN * 4);
  float* inv4 = (float*)carve((size_t)BATCH * 4);
  float* agg4 = (float*)carve((size_t)32 * 3072 * 4);
  float* x4   = (float*)carve((size_t)32 * 384 * 4);
  float* x4n  = (float*)carve((size_t)32 * 384 * 4);

  auto gemm_blocks = [](int M, int N) {
    const int waves = ((M + 15) / 16) * ((N + 15) / 16);
    return (waves + 7) / 8;
  };

  // ---- layer 1: n_out=128, neighbors among 16384, C=3 ----
  knn_kernel<<<BATCH, 128, 0, stream>>>(pts, NPTS, 128, idx1, inv1);
  geo_mlp_kernel<<<(BATCH * 128) / 8, 256, 0, stream>>>(
      pts, x, idx1, inv1, ws1_1, bs1_1, ws2_1, agg1, 128, NPTS, 3);
  gemm_kernel<<<gemm_blocks(4096, 64), 256, 0, stream>>>(
      agg1, wsem1, bsem1, x1, 4096, 64, 48, /*leaky=*/1, /*leaky_a=*/0);

  // ---- layer 3: n_out=32, neighbors among first 128 pts, C=64 ----
  knn_kernel<<<BATCH, 128, 0, stream>>>(pts, 128, 32, idx3, inv3);
  geo_mlp_kernel<<<(BATCH * 32) / 8, 256, 0, stream>>>(
      pts, x1, idx3, inv3, ws1_3, bs1_3, ws2_3, agg3, 32, 128, 64);
  gemm_kernel<<<gemm_blocks(1024, 192), 256, 0, stream>>>(
      agg3, wsem3, bsem3, x3, 1024, 192, 1024, 1, 0);

  // ---- layer 4: n_out=1, neighbors among first 32 pts, C=192 ----
  knn_kernel<<<BATCH, 128, 0, stream>>>(pts, 32, 1, idx4, inv4);
  geo_mlp_kernel<<<(BATCH * 1) / 8, 256, 0, stream>>>(
      pts, x3, idx4, inv4, ws1_4, bs1_4, ws2_4, agg4, 1, 32, 192);
  gemm_kernel<<<gemm_blocks(32, 384), 256, 0, stream>>>(
      agg4, wsem4, bsem4, x4, 32, 384, 3072, /*leaky=*/0, 0);

  // ---- batchnorm + leaky ----
  bn_leaky_kernel<<<(384 + 255) / 256, 256, 0, stream>>>(x4, x4n, 32, 384);

  // ---- xout = x4n @ w_fcout  (raw, first output) ----
  gemm_kernel<<<gemm_blocks(32, 128), 256, 0, stream>>>(
      x4n, wfco, nullptr, out, 32, 128, 384, 0, 0);
  // ---- xreg = leaky(xout) @ w_fcout2 (second output) ----
  gemm_kernel<<<gemm_blocks(32, 20), 256, 0, stream>>>(
      out, wfco2, nullptr, out + 4096, 32, 20, 128, 0, /*leaky_a=*/1);

  (void)in_sizes; (void)n_in; (void)out_size; (void)ws_size;
}
